// GraphPointTransformer_77841987272928
// MI455X (gfx1250) — matrix operand order repaired and checked by reading, not compiled, not run on hardware
//
#include <hip/hip_runtime.h>
#include <math.h>

// ---------------------------------------------------------------------------
// Types / constants
// ---------------------------------------------------------------------------
typedef _Float16 v16h __attribute__((ext_vector_type(16)));
typedef _Float16 v8h  __attribute__((ext_vector_type(8)));
typedef float    v8f  __attribute__((ext_vector_type(8)));
typedef float    v4f  __attribute__((ext_vector_type(4)));

#define ACT_NONE  0
#define ACT_RELU  1
#define ACT_LRELU 2

#define TM 64
#define TN 64
#define TK 32
// 40 halves = 80 bytes per row -> every row and every 8-half group 16B-aligned
#define AS_STRIDE 40   // A tile: 64 rows (M) x 32 (K), row-major
#define BS_STRIDE 40   // B tile stored TRANSPOSED: 64 rows (N) x 32 (K)

static const float BN_SCALE = 0.9999950000374997f; // 1/sqrt(1+1e-5)

// ---------------------------------------------------------------------------
// WMMA GEMM:  C[M,N] = act(scale*(A[M,K] @ B[K,N] + bias)) + res
// ---------------------------------------------------------------------------
__device__ __forceinline__ float act_apply(float v, int act) {
  if (act == ACT_RELU)  return v > 0.0f ? v : 0.0f;
  if (act == ACT_LRELU) return v > 0.0f ? v : 0.01f * v;
  return v;
}

__device__ __forceinline__ v16h cat8(v8h lo, v8h hi) {
  return __builtin_shufflevector(lo, hi, 0, 1, 2, 3, 4, 5, 6, 7,
                                 8, 9, 10, 11, 12, 13, 14, 15);
}

// A fragment (16x32 f16): lane half h=(lane>>4): K = h*8 + {0..7} and 16+h*8+{0..7}
// -> two aligned 16B LDS loads from the row-major A tile.
__device__ __forceinline__ v16h frag_a(const _Float16* As, int mBase, int lane) {
  const int m  = mBase + (lane & 15);
  const int kh = (lane >> 4) * 8;
  const _Float16* r = As + m * AS_STRIDE;
  const v8h lo = *(const v8h*)(r + kh);
  const v8h hi = *(const v8h*)(r + 16 + kh);
  return cat8(lo, hi);
}

// B fragment (32x16 f16): column N = lane&15, K = h*8 + {0..7} and 16+h*8+{0..7}
// (same K order as A) -> in the transposed tile:
// two aligned 16B LDS loads.
__device__ __forceinline__ v16h frag_b(const _Float16* Bs, int nBase, int lane) {
  const int n  = nBase + (lane & 15);
  const int kh = (lane >> 4) * 8;
  const _Float16* r = Bs + n * BS_STRIDE + kh;
  const v8h lo = *(const v8h*)(r);
  const v8h hi = *(const v8h*)(r + 16);
  return cat8(lo, hi);
}

__global__ __launch_bounds__(256) void gemm_wmma_kernel(
    const float* __restrict__ A, const float* __restrict__ B,
    const float* __restrict__ bias, const float* __restrict__ res,
    float* __restrict__ C, int M, int N, int K, float scale, int act)
{
  __shared__ _Float16 As[TM * AS_STRIDE];
  __shared__ _Float16 Bs[TN * BS_STRIDE];
  const int tid  = threadIdx.x;
  const int lane = tid & 31;
  const int wid  = tid >> 5;           // 8 waves
  const int wm   = wid >> 1;           // 0..3  (16-row tile)
  const int wn   = wid & 1;            // 0..1  (32-col slab -> 2 wmma)
  const int m0   = blockIdx.y * TM;
  const int n0   = blockIdx.x * TN;

  // A staging map: row r = tid/4 (0..63), 8 K's starting at (tid&3)*8
  const int ar  = tid >> 2;
  const int ac0 = (tid & 3) * 8;
  // B staging map: col n = tid&63, 8 K's starting at (tid>>6)*8
  const int bn  = tid & 63;
  const int bk0 = (tid >> 6) * 8;

  v8f acc0 = {};
  v8f acc1 = {};

  for (int k0 = 0; k0 < K; k0 += TK) {
    // ---- stage A (f32 -> f16), one ds_store_b128 per thread ----
    {
      const int gm = m0 + ar;
      v8h h;
      if (m0 + TM <= M && k0 + TK <= K && (K & 3) == 0) {
        // uniform fast path: two aligned float4 global loads
        const float* src = A + (size_t)gm * K + k0 + ac0;
        const v4f f0 = *(const v4f*)(src);
        const v4f f1 = *(const v4f*)(src + 4);
#pragma unroll
        for (int u = 0; u < 4; ++u) { h[u] = (_Float16)f0[u]; h[4 + u] = (_Float16)f1[u]; }
      } else {
        // branch-free guarded path (clamped addresses, select to zero)
        const int cm = gm < M ? gm : M - 1;
#pragma unroll
        for (int u = 0; u < 8; ++u) {
          const int gk = k0 + ac0 + u;
          const int ck = gk < K ? gk : K - 1;
          float v = A[(size_t)cm * K + ck];
          if (gm >= M || gk >= K) v = 0.0f;
          h[u] = (_Float16)v;
        }
      }
      *(v8h*)(&As[ar * AS_STRIDE + ac0]) = h;
    }
    // ---- stage B transposed (Bs[n][k]), one ds_store_b128 per thread ----
    {
      const int gn = n0 + bn;
      v8h h;
      if (n0 + TN <= N && k0 + TK <= K) {
#pragma unroll
        for (int u = 0; u < 8; ++u)
          h[u] = (_Float16)B[(size_t)(k0 + bk0 + u) * N + gn];
      } else {
        const int cn = gn < N ? gn : N - 1;
#pragma unroll
        for (int u = 0; u < 8; ++u) {
          const int gk = k0 + bk0 + u;
          const int ck = gk < K ? gk : K - 1;
          float v = B[(size_t)ck * N + cn];
          if (gn >= N || gk >= K) v = 0.0f;
          h[u] = (_Float16)v;
        }
      }
      *(v8h*)(&Bs[bn * BS_STRIDE + bk0]) = h;
    }
    // CDNA5 global_prefetch_b8 of next K tiles
    if (k0 + TK < K) {
      const int gm = m0 + ar;
      if (gm < M) __builtin_prefetch(&A[(size_t)gm * K + (k0 + TK)], 0, 1);
      if (k0 + TK + bk0 < K)
        __builtin_prefetch(&B[(size_t)(k0 + TK + bk0) * N + n0], 0, 1);
    }
    __syncthreads();

    const v16h a  = frag_a(As, wm * 16, lane);
    const v16h b0 = frag_b(Bs, wn * 32, lane);
    const v16h b1 = frag_b(Bs, wn * 32 + 16, lane);
    acc0 = __builtin_amdgcn_wmma_f32_16x16x32_f16(false, a, false, b0, (short)0, acc0, false, false);
    acc1 = __builtin_amdgcn_wmma_f32_16x16x32_f16(false, a, false, b1, (short)0, acc1, false, false);
    __syncthreads();
  }

  // C/D layout: VGPR r -> M = r (lanes 0-15) / r+8 (lanes 16-31); N = lane&15
  const int mh = (lane >> 4) * 8;
  const int nl = lane & 15;
#pragma unroll
  for (int r = 0; r < 8; ++r) {
    const int gm = m0 + wm * 16 + mh + r;
    if (gm >= M) continue;
    int gn = n0 + wn * 32 + nl;
    if (gn < N) {
      float v = acc0[r] + (bias ? bias[gn] : 0.0f);
      v = act_apply(v * scale, act);
      if (res) v += res[(size_t)gm * N + gn];
      C[(size_t)gm * N + gn] = v;
    }
    gn += 16;
    if (gn < N) {
      float v = acc1[r] + (bias ? bias[gn] : 0.0f);
      v = act_apply(v * scale, act);
      if (res) v += res[(size_t)gm * N + gn];
      C[(size_t)gm * N + gn] = v;
    }
  }
}

// ---------------------------------------------------------------------------
// kNN (top-KN smallest sq-dist, insertion sort, first-index tie break)
// ---------------------------------------------------------------------------
template <int KN>
__global__ void knn_kernel(const float* __restrict__ q, const float* __restrict__ base,
                           int* __restrict__ out, int Mq, int Nb, int excludeSelf)
{
  const int i = blockIdx.x * blockDim.x + threadIdx.x;
  if (i >= Mq) return;
  const float qx = q[3 * i + 0], qy = q[3 * i + 1], qz = q[3 * i + 2];
  float bd[KN];
  int   bi[KN];
#pragma unroll
  for (int t = 0; t < KN; ++t) { bd[t] = 3.4e38f; bi[t] = 0; }
  for (int j = 0; j < Nb; ++j) {
    if (excludeSelf && j == i) continue;
    const float dx = base[3 * j + 0] - qx;
    const float dy = base[3 * j + 1] - qy;
    const float dz = base[3 * j + 2] - qz;
    const float d  = dx * dx + dy * dy + dz * dz;
    if (d < bd[KN - 1]) {
      int t = KN - 1;
      while (t > 0 && bd[t - 1] > d) { bd[t] = bd[t - 1]; bi[t] = bi[t - 1]; --t; }
      bd[t] = d; bi[t] = j;
    }
  }
#pragma unroll
  for (int t = 0; t < KN; ++t) out[(size_t)i * KN + t] = bi[t];
}

// ---------------------------------------------------------------------------
// Farthest point sampling (single block, sequential scan like the reference)
// ---------------------------------------------------------------------------
__global__ __launch_bounds__(1024) void fps_kernel(const float* __restrict__ pos, int n, int m,
                                                   int* __restrict__ out, float* __restrict__ mind)
{
  __shared__ float sval[1024];
  __shared__ int   sidx[1024];
  const int t = threadIdx.x;
  for (int j = t; j < n; j += 1024) mind[j] = 3.4e38f;
  if (t == 0) out[0] = 0;
  __syncthreads();
  int last = 0;
  for (int s = 1; s < m; ++s) {
    const float px = pos[3 * last + 0], py = pos[3 * last + 1], pz = pos[3 * last + 2];
    float best = -1.0f; int bestj = 0x7fffffff;
    for (int j = t; j < n; j += 1024) {
      const float dx = pos[3 * j + 0] - px;
      const float dy = pos[3 * j + 1] - py;
      const float dz = pos[3 * j + 2] - pz;
      const float d  = dx * dx + dy * dy + dz * dz;
      const float md = fminf(mind[j], d);
      mind[j] = md;
      if (md > best || (md == best && j < bestj)) { best = md; bestj = j; }
    }
    sval[t] = best; sidx[t] = bestj;
    __syncthreads();
    for (int off = 512; off > 0; off >>= 1) {
      if (t < off) {
        if (sval[t + off] > sval[t] ||
            (sval[t + off] == sval[t] && sidx[t + off] < sidx[t])) {
          sval[t] = sval[t + off]; sidx[t] = sidx[t + off];
        }
      }
      __syncthreads();
    }
    last = sidx[0];
    if (t == 0) out[s] = last;
    __syncthreads();
  }
}

// ---------------------------------------------------------------------------
// Elementwise / gather kernels
// ---------------------------------------------------------------------------
__global__ void gather_rows_kernel(const float* __restrict__ in, const int* __restrict__ idx,
                                   float* __restrict__ out, int M, int C)
{
  const int g = blockIdx.x * blockDim.x + threadIdx.x;
  if (g >= M * C) return;
  const int i = g / C, c = g - i * C;
  out[g] = in[(size_t)idx[i] * C + c];
}

__global__ void gather_max_kernel(const float* __restrict__ x, const int* __restrict__ nb,
                                  float* __restrict__ out, int M, int C)
{
  const int g = blockIdx.x * blockDim.x + threadIdx.x;
  if (g >= M * C) return;
  const int i = g / C, c = g - i * C;
  float m = -3.4e38f;
#pragma unroll
  for (int j = 0; j < 16; ++j) m = fmaxf(m, x[(size_t)nb[i * 16 + j] * C + c]);
  out[g] = m;
}

// pos diffs over [N, 17, 3] edges (17th neighbor = self loop, diff = 0)
__global__ void posdiff_kernel(const float* __restrict__ pos, const int* __restrict__ nb,
                               float* __restrict__ pd, int N)
{
  const int g = blockIdx.x * blockDim.x + threadIdx.x;
  if (g >= N * 17 * 3) return;
  const int d = g % 3;
  const int e = g / 3;
  const int j = e % 17;
  const int i = e / 17;
  const int src = (j == 16) ? i : nb[i * 16 + j];
  pd[g] = pos[3 * i + d] - pos[3 * src + d];
}

// a_dst[i] - a_src[nb] + delta  over [N, 17, C]
__global__ void attn_in_kernel(const float* __restrict__ adst, const float* __restrict__ asrc,
                               const float* __restrict__ delta, const int* __restrict__ nb,
                               float* __restrict__ out, int N, int C)
{
  const int g = blockIdx.x * blockDim.x + threadIdx.x;
  if (g >= N * 17 * C) return;
  const int c = g % C;
  const int e = g / C;
  const int j = e % 17;
  const int i = e / 17;
  const int src = (j == 16) ? i : nb[i * 16 + j];
  out[g] = adst[(size_t)i * C + c] - asrc[(size_t)src * C + c] + delta[g];
}

// softmax over neighbor axis (17) for each (i, c), in place on [N,17,C]
__global__ void softmax_nb_kernel(float* __restrict__ a, int N, int C)
{
  const int g = blockIdx.x * blockDim.x + threadIdx.x;
  if (g >= N * C) return;
  const int i = g / C, c = g - i * C;
  const size_t base = (size_t)i * 17 * C + c;
  float mx = -3.4e38f;
  for (int j = 0; j < 17; ++j) mx = fmaxf(mx, a[base + (size_t)j * C]);
  float s = 0.0f;
  for (int j = 0; j < 17; ++j) {
    const float e = __expf(a[base + (size_t)j * C] - mx);
    a[base + (size_t)j * C] = e;
    s += e;
  }
  const float inv = 1.0f / s;
  for (int j = 0; j < 17; ++j) a[base + (size_t)j * C] *= inv;
}

// out[i,c] = sum_j alpha[i,j,c] * (xs[nb[i,j],c] + delta[i,j,c])
__global__ void aggregate_kernel(const float* __restrict__ alpha, const float* __restrict__ xs,
                                 const float* __restrict__ delta, const int* __restrict__ nb,
                                 float* __restrict__ out, int N, int C)
{
  const int g = blockIdx.x * blockDim.x + threadIdx.x;
  if (g >= N * C) return;
  const int i = g / C, c = g - i * C;
  float s = 0.0f;
  for (int j = 0; j < 17; ++j) {
    const int src = (j == 16) ? i : nb[i * 16 + j];
    const size_t e = ((size_t)i * 17 + j) * C + c;
    s += alpha[e] * (xs[(size_t)src * C + c] + delta[e]);
  }
  out[g] = s;
}

// knn_interpolate, k=3, inverse-square-distance weights
__global__ void interp_kernel(const float* __restrict__ xsub, const int* __restrict__ idx,
                              const float* __restrict__ pos, const float* __restrict__ possub,
                              float* __restrict__ xi, int Nq, int C)
{
  const int g = blockIdx.x * blockDim.x + threadIdx.x;
  if (g >= Nq * C) return;
  const int i = g / C, c = g - i * C;
  const float px = pos[3 * i + 0], py = pos[3 * i + 1], pz = pos[3 * i + 2];
  float num = 0.0f, den = 0.0f;
#pragma unroll
  for (int t = 0; t < 3; ++t) {
    const int j = idx[i * 3 + t];
    const float dx = px - possub[3 * j + 0];
    const float dy = py - possub[3 * j + 1];
    const float dz = pz - possub[3 * j + 2];
    const float d2 = fmaxf(dx * dx + dy * dy + dz * dz, 1e-16f);
    const float w  = 1.0f / d2;
    num += w * xsub[(size_t)j * C + c];
    den += w;
  }
  xi[g] = num / den;
}

__global__ void softmax_row8_kernel(const float* __restrict__ in, float* __restrict__ out, int N)
{
  const int i = blockIdx.x * blockDim.x + threadIdx.x;
  if (i >= N) return;
  float v[8], mx = -3.4e38f;
#pragma unroll
  for (int c = 0; c < 8; ++c) { v[c] = in[(size_t)i * 8 + c]; mx = fmaxf(mx, v[c]); }
  float s = 0.0f;
#pragma unroll
  for (int c = 0; c < 8; ++c) { v[c] = __expf(v[c] - mx); s += v[c]; }
  const float inv = 1.0f / s;
#pragma unroll
  for (int c = 0; c < 8; ++c) out[(size_t)i * 8 + c] = v[c] * inv;
}

// ---------------------------------------------------------------------------
// Host orchestration
// ---------------------------------------------------------------------------
static inline const float* F(void* const* din, int i) { return (const float*)din[i]; }
static inline dim3 g1(size_t n) { return dim3((unsigned)((n + 255) / 256)); }

static void gemm(hipStream_t s, const float* A, const float* B, const float* bias,
                 const float* res, float* C, int M, int N, int K, float scale, int act)
{
  dim3 grid((N + TN - 1) / TN, (M + TM - 1) / TM);
  gemm_wmma_kernel<<<grid, dim3(256), 0, s>>>(A, B, bias, res, C, M, N, K, scale, act);
}

// Block leaf offsets (pytree sorted-key order within a _block_params dict):
// +0 attn0.W +1 attn0.b +2 attn1.W +3 attn1.b +4 lin.W +5 lin_dst.W
// +6 lin_in.W +7 lin_in.b +8 lin_out.W +9 lin_out.b +10 lin_src.W
// +11 pos0.W +12 pos0.b +13 pos1.W +14 pos1.b
static void run_block(hipStream_t s, void* const* din, int pb,
                      const float* xin, const float* pos, const int* nb,
                      float* xout, int N, int C, char* arena)
{
  char* p = arena;
  auto aF = [&](size_t n) { float* r = (float*)p; p += ((n * 4 + 255) / 256) * 256; return r; };
  const int E = N * 17;
  float* xh    = aF((size_t)N * C);
  float* xs    = aF((size_t)N * C);
  float* asrc  = aF((size_t)N * C);
  float* adst  = aF((size_t)N * C);
  float* agg   = aF((size_t)N * C);
  float* pd    = aF((size_t)E * 3);
  float* hid   = aF((size_t)E * 64);
  float* delta = aF((size_t)E * C);
  float* ebuf  = aF((size_t)E * C);  // attn-in, then alpha (in place)

  gemm(s, xin, F(din, pb + 6), F(din, pb + 7), nullptr, xh, N, C, C, 1.0f, ACT_RELU); // lin_in
  gemm(s, xh, F(din, pb + 4),  nullptr, nullptr, xs,   N, C, C, 1.0f, ACT_NONE);      // lin
  gemm(s, xh, F(din, pb + 10), nullptr, nullptr, asrc, N, C, C, 1.0f, ACT_NONE);      // lin_src
  gemm(s, xh, F(din, pb + 5),  nullptr, nullptr, adst, N, C, C, 1.0f, ACT_NONE);      // lin_dst

  posdiff_kernel<<<g1((size_t)N * 17 * 3), 256, 0, s>>>(pos, nb, pd, N);
  gemm(s, pd,  F(din, pb + 11), F(din, pb + 12), nullptr, hid,   E, 64, 3,  1.0f, ACT_LRELU); // pos_nn[0]
  gemm(s, hid, F(din, pb + 13), F(din, pb + 14), nullptr, delta, E, C,  64, 1.0f, ACT_LRELU); // pos_nn[1]

  attn_in_kernel<<<g1((size_t)E * C), 256, 0, s>>>(adst, asrc, delta, nb, ebuf, N, C);
  gemm(s, ebuf, F(din, pb + 0), F(din, pb + 1), nullptr, hid,  E, 64, C,  1.0f, ACT_RELU);    // attn_nn[0]
  gemm(s, hid,  F(din, pb + 2), F(din, pb + 3), nullptr, ebuf, E, C,  64, 1.0f, ACT_RELU);    // attn_nn[1]
  softmax_nb_kernel<<<g1((size_t)N * C), 256, 0, s>>>(ebuf, N, C);
  aggregate_kernel<<<g1((size_t)N * C), 256, 0, s>>>(ebuf, xs, delta, nb, agg, N, C);
  gemm(s, agg, F(din, pb + 8), F(din, pb + 9), nullptr, xout, N, C, C, 1.0f, ACT_RELU);       // lin_out
}

extern "C" void kernel_launch(void* const* d_in, const int* in_sizes, int n_in,
                              void* d_out, int out_size, void* d_ws, size_t ws_size,
                              hipStream_t stream)
{
  (void)in_sizes; (void)n_in; (void)out_size; (void)ws_size;
  const float* x_in = F(d_in, 0);  // [8192,4]
  const float* pos0 = F(d_in, 1);  // [8192,3]

  // ---- parameter leaf indexing (sorted-key pytree flatten of params) ----
  int o = 2;
  const int ibSummit = o; o += 15;
  int dblk[3], dtdW[3], dtdB[3];
  for (int i = 0; i < 3; ++i) { dblk[i] = o; dtdW[i] = o + 15; dtdB[i] = o + 16; o += 17; }
  const int iInW = o, iInB = o + 1; o += 2;
  const int iO0W = o, iO0B = o + 1, iO1W = o + 2, iO1B = o + 3; o += 4;
  const int iSmW = o, iSmB = o + 1; o += 2;
  const int ibTin = o; o += 15;
  int ublk[3], umW[3], umB[3], usW[3], usB[3];
  for (int i = 0; i < 3; ++i) {
    ublk[i] = o; umW[i] = o + 15; umB[i] = o + 16; usW[i] = o + 17; usB[i] = o + 18; o += 19;
  }

  const int NL[4] = {8192, 2048, 512, 128};
  const int CL[4] = {32, 128, 512, 2048};

  // ---- workspace layout (deterministic bump allocator) ----
  char* p = (char*)d_ws;
  auto aF = [&](size_t n) { float* r = (float*)p; p += ((n * 4 + 255) / 256) * 256; return r; };
  auto aI = [&](size_t n) { int* r = (int*)p;   p += ((n * 4 + 255) / 256) * 256; return r; };

  float* xa = aF((size_t)8192 * 32);
  float* xL[4];
  for (int l = 0; l < 4; ++l) xL[l] = aF((size_t)NL[l] * CL[l]);
  const float* posL[4];
  posL[0] = pos0;
  float* posBuf[4] = {nullptr, aF(2048 * 3), aF(512 * 3), aF(128 * 3)};
  for (int l = 1; l < 4; ++l) posL[l] = posBuf[l];
  int* nbrL[4];
  for (int l = 0; l < 4; ++l) nbrL[l] = aI((size_t)NL[l] * 16);
  int* nbd     = aI((size_t)2048 * 16);
  int* knn3    = aI((size_t)8192 * 3);
  int* fpsIdx  = aI(2048);
  float* mind  = aF(8192);
  float* tdx   = aF((size_t)8192 * 128);
  float* xdown = aF((size_t)2048 * 128);
  float* smv   = aF((size_t)128 * 2048);
  float* tsub  = aF(65536);
  float* xi    = aF(262144);
  float* cur   = aF(262144);
  float* cur2  = aF(262144);
  float* h64   = aF((size_t)8192 * 64);
  float* logit = aF((size_t)8192 * 8);
  char* arena  = p;

  // ---- encoder ----
  gemm(stream, x_in, F(d_in, iInW), F(d_in, iInB), nullptr, xa, 8192, 32, 4, BN_SCALE, ACT_RELU);
  knn_kernel<16><<<g1(8192), 256, 0, stream>>>(pos0, pos0, nbrL[0], 8192, 8192, 1);
  run_block(stream, d_in, ibTin, xa, posL[0], nbrL[0], xL[0], 8192, 32, arena);

  for (int l = 0; l < 3; ++l) {
    const int n = NL[l], m = NL[l + 1];
    fps_kernel<<<1, 1024, 0, stream>>>(posL[l], n, m, fpsIdx, mind);
    gather_rows_kernel<<<g1((size_t)m * 3), 256, 0, stream>>>(posL[l], fpsIdx, posBuf[l + 1], m, 3);
    knn_kernel<16><<<g1(m), 256, 0, stream>>>(posL[l + 1], posL[l], nbd, m, n, 0);
    gemm(stream, xL[l], F(d_in, dtdW[l]), F(d_in, dtdB[l]), nullptr, tdx,
         n, CL[l + 1], CL[l], BN_SCALE, ACT_RELU);
    gather_max_kernel<<<g1((size_t)m * CL[l + 1]), 256, 0, stream>>>(tdx, nbd, xdown, m, CL[l + 1]);
    knn_kernel<16><<<g1(m), 256, 0, stream>>>(posL[l + 1], posL[l + 1], nbrL[l + 1], m, m, 1);
    run_block(stream, d_in, dblk[l], xdown, posL[l + 1], nbrL[l + 1], xL[l + 1], m, CL[l + 1], arena);
  }

  // ---- summit ----
  gemm(stream, xL[3], F(d_in, iSmW), F(d_in, iSmB), nullptr, smv, 128, 2048, 2048, 1.0f, ACT_RELU);
  run_block(stream, d_in, ibSummit, smv, posL[3], nbrL[3], cur, 128, 2048, arena);

  // ---- decoder ----
  for (int step = 0; step < 3; ++step) {
    const int lh = 3 - step;        // fine-to-coarse source level
    const int ll = 2 - step;        // target level
    const int ui = ll;              // up[ll]: DIMS[ll+1] -> DIMS[ll]
    gemm(stream, cur, F(d_in, usW[ui]), F(d_in, usB[ui]), nullptr, tsub,
         NL[lh], CL[ll], CL[lh], BN_SCALE, ACT_RELU);
    knn_kernel<3><<<g1(NL[ll]), 256, 0, stream>>>(posL[ll], posL[lh], knn3, NL[ll], NL[lh], 0);
    interp_kernel<<<g1((size_t)NL[ll] * CL[ll]), 256, 0, stream>>>(
        tsub, knn3, posL[ll], posL[lh], xi, NL[ll], CL[ll]);
    gemm(stream, xL[ll], F(d_in, umW[ui]), F(d_in, umB[ui]), xi, cur2,
         NL[ll], CL[ll], CL[ll], BN_SCALE, ACT_RELU);
    run_block(stream, d_in, ublk[ui], cur2, posL[ll], nbrL[ll], cur, NL[ll], CL[ll], arena);
  }

  // ---- head: 32 -> 64 relu -> 8 softmax ----
  gemm(stream, cur, F(d_in, iO0W), F(d_in, iO0B), nullptr, h64, 8192, 64, 32, 1.0f, ACT_RELU);
  gemm(stream, h64, F(d_in, iO1W), F(d_in, iO1B), nullptr, logit, 8192, 8, 64, 1.0f, ACT_NONE);
  softmax_row8_kernel<<<g1(8192), 256, 0, stream>>>(logit, (float*)d_out, 8192);
}
